// ESN_5248450035747
// MI455X (gfx1250) — matrix operand order repaired and checked by reading, not compiled, not run on hardware
//
#include <hip/hip_runtime.h>
#include <hip/hip_bf16.h>
#include <math.h>

typedef __attribute__((ext_vector_type(16))) __bf16 v16bf;
typedef __attribute__((ext_vector_type(8)))  float  v8f;
typedef __attribute__((ext_vector_type(4))) unsigned int u32x4;
typedef __attribute__((ext_vector_type(8))) int i32x8;
typedef __attribute__((ext_vector_type(4))) int i32x4;

#define T_STEPS   8192
#define IN_DIM    32
#define RES       2048
#define NWG       32          // persistent cooperative workgroups
#define TPB       128         // 4 waves (wave32)
#define WAVES     4
#define ROWS_WG   64          // RES / NWG
#define KCHUNKS   64          // RES / 32

// ---- dynamic LDS layout (bytes) ----
#define LDS_W_HALF     (ROWS_WG * RES)            // 131072 bf16 = 256KB, WMMA A-frag order
#define LDS_H_OFF_B    (LDS_W_HALF * 2)           // 262144 : h vector, bf16[2048]
#define LDS_WOUT_OFF_B (LDS_H_OFF_B + RES * 2)    // 266240 : W_out f32[2048] (WG0 only)
#define LDS_RED_OFF_B  (LDS_WOUT_OFF_B + RES * 4) // 274432 : reduction scratch
#define LDS_ZERO_OFF_B (LDS_RED_OFF_B + 64)       // 274496 : 64B zero block for B frags
#define LDS_TOTAL_B    (LDS_ZERO_OFF_B + 64)      // 274560

// ---- workspace layout (bytes) ----
#define WS_CNT_OFF   0        // unsigned arrive, release
#define WS_HPING_OFF 128      // bf16[2048]
#define WS_HPONG_OFF (128 + 4096)

// ---------------------------------------------------------------------------
// init: reset barrier counters, seed h0 (bf16) into the PONG buffer
// ---------------------------------------------------------------------------
__global__ void esn_init_kernel(const float* __restrict__ h0,
                                __bf16* __restrict__ hping,
                                __bf16* __restrict__ hpong,
                                unsigned* __restrict__ counters) {
    int i = blockIdx.x * blockDim.x + threadIdx.x;
    if (i < RES) {
        hpong[i] = (__bf16)h0[i];
        hping[i] = (__bf16)0.0f;
    }
    if (i == 0) { counters[0] = 0u; counters[1] = 0u; }
}

// ---------------------------------------------------------------------------
// TDM: pull the full bf16 h vector (4KB) from global into LDS
// ---------------------------------------------------------------------------
__device__ __forceinline__ void tdm_load_h(const __bf16* gsrc, __bf16* ldsH) {
#if __has_builtin(__builtin_amdgcn_tensor_load_to_lds)
    unsigned long long ga = (unsigned long long)(uintptr_t)gsrc;
    u32x4 g0;
    g0[0] = 1u;                                            // count=1, user mode
    g0[1] = (unsigned)LDS_H_OFF_B;                         // lds_addr
    g0[2] = (unsigned)(ga & 0xFFFFFFFFull);                // global_addr[31:0]
    g0[3] = (unsigned)((ga >> 32) & 0x1FFFFFFull) | (2u << 30); // addr[56:32] | type=2
    i32x8 g1;
    g1[0] = (1 << 16);                                     // data_size = 2B
    g1[1] = (int)((RES & 0xFFFF) << 16);                   // tensor_dim0 lo
    g1[2] = (int)((RES >> 16) & 0xFFFF) | (1 << 16);       // dim0 hi | tensor_dim1=1
    g1[3] = (int)(RES << 16);                              // tile_dim0 = 2048
    g1[4] = 1;                                             // tile_dim1 = 1
    g1[5] = RES;                                           // tensor_dim0_stride lo32
    g1[6] = 0; g1[7] = 0;
    i32x4 g2 = {0, 0, 0, 0};
    i32x4 g3 = {0, 0, 0, 0};
    i32x8 g4 = {0, 0, 0, 0, 0, 0, 0, 0};
    __builtin_amdgcn_tensor_load_to_lds(g0, g1, g2, g3, g4, 0);
    __builtin_amdgcn_s_wait_tensorcnt(0);
#else
    int l = threadIdx.x & 31;
    const uint4* s = (const uint4*)gsrc;
    uint4* d = (uint4*)ldsH;
    for (int i = l; i < (RES * 2) / 16; i += 32) d[i] = s[i];
#endif
}

// ---------------------------------------------------------------------------
// grid-wide sense barrier (persistent cooperative kernel)
// ---------------------------------------------------------------------------
__device__ __forceinline__ void grid_sync(unsigned* arrive, unsigned* release,
                                          unsigned gen) {
    __syncthreads();
    if (threadIdx.x == 0) {
        __threadfence();
        unsigned old = __hip_atomic_fetch_add(arrive, 1u, __ATOMIC_ACQ_REL,
                                              __HIP_MEMORY_SCOPE_AGENT);
        if (old == (unsigned)(NWG - 1)) {
            __hip_atomic_store(arrive, 0u, __ATOMIC_RELAXED, __HIP_MEMORY_SCOPE_AGENT);
            __hip_atomic_store(release, gen, __ATOMIC_RELEASE, __HIP_MEMORY_SCOPE_AGENT);
        } else {
            while (__hip_atomic_load(release, __ATOMIC_ACQUIRE,
                                     __HIP_MEMORY_SCOPE_AGENT) < gen) {
                __builtin_amdgcn_s_sleep(1);
            }
        }
    }
    __syncthreads();
}

// ---------------------------------------------------------------------------
// main persistent ESN kernel
// ---------------------------------------------------------------------------
__global__ void __launch_bounds__(TPB, 1)
esn_persistent_kernel(const float* __restrict__ u,
                      const float* __restrict__ W_in,
                      const float* __restrict__ W,
                      const float* __restrict__ W_out,
                      const float* __restrict__ b_out,
                      float* __restrict__ out,
                      __bf16* __restrict__ hping,
                      __bf16* __restrict__ hpong,
                      unsigned* __restrict__ counters) {
    extern __shared__ char smem[];
    __bf16* ldsW    = (__bf16*)smem;
    __bf16* ldsH    = (__bf16*)(smem + LDS_H_OFF_B);
    float*  ldsWout = (float*)(smem + LDS_WOUT_OFF_B);
    float*  ldsRed  = (float*)(smem + LDS_RED_OFF_B);
    __bf16* zeroLds = (__bf16*)(smem + LDS_ZERO_OFF_B);

    const int tid   = threadIdx.x;
    const int lane  = tid & 31;
    const int wave  = tid >> 5;
    const int wg    = blockIdx.x;
    const int wgRow0 = wg * ROWS_WG;

    unsigned* arrive  = counters + 0;
    unsigned* release = counters + 1;

    // ---- one-time: load + bf16-convert + A-fragment-swizzle our W slice ----
    {
        const float* Wslice = W + (size_t)wgRow0 * RES;
        for (int idx = tid; idx < ROWS_WG * RES; idx += TPB) {
            int r = idx >> 11;           // row in slice (0..63)
            int j = idx & (RES - 1);     // column (0..2047)
            float w = Wslice[(size_t)r * RES + j];
            int frag = (r >> 4) * KCHUNKS + (j >> 5);
            int fl   = (r & 15) + (((j >> 4) & 1) << 4);       // lane in fragment
            int slot = j & 15;                                 // bf16 slot in lane
            ldsW[frag * 512 + fl * 16 + slot] = (__bf16)w;
        }
    }
    // W_out cached in WG0's LDS for the readout
    if (wg == 0) {
        for (int j = tid; j < RES; j += TPB) ldsWout[j] = W_out[j];
    }
    if (tid < 32) zeroLds[tid] = (__bf16)0.0f;

    // ---- W_in A-fragment: one 16x32 tile per wave, resident in VGPRs ----
    const int m     = lane & 15;
    const int halfA = lane >> 4;
    const int inCol0 = ((lane & 15) == 0);
    const int rowBase = wgRow0 + wave * 16;
    v16bf aIn;
    #pragma unroll
    for (int s = 0; s < 16; ++s) {
        int kk = (halfA << 4) | s;
        aIn[s] = (__bf16)W_in[(size_t)(rowBase + m) * IN_DIM + kk];
    }

    // B-fragment source: lanes 0/16 read h chunks, all others read zeros
    const __bf16* hSrc   = inCol0 ? (ldsH + halfA * 16) : zeroLds;
    const int     hStride = inCol0 ? 32 : 0;

    const float bo = b_out[0];
    const __bf16* myW = ldsW + (unsigned)(wave * KCHUNKS) * 512u + (unsigned)lane * 16u;

    // ---- initial h_{-1} = h0 from PONG buffer via TDM ----
    if (wave == 0) tdm_load_h(hpong, ldsH);
    __syncthreads();

    // =================== sequential recurrence ===================
    for (int t = 0; t < T_STEPS; ++t) {
        if (tid == 0 && t + 1 < T_STEPS)
            __builtin_prefetch(u + (size_t)(t + 1) * IN_DIM, 0, 1);

        // C init: W_in @ u_t  (u_t as single-column B fragment)
        v16bf bu = {};
        if (inCol0) {
            const float* up = u + (size_t)t * IN_DIM + halfA * 16;
            #pragma unroll
            for (int s = 0; s < 16; ++s) bu[s] = (__bf16)up[s];
        }
        v8f c = {};
        c = __builtin_amdgcn_wmma_f32_16x16x32_bf16(false, aIn, false, bu,
                                                    (short)0, c, false, false);

        // accumulate W @ h_{t-1}: 64 chained WMMAs, K = 2048.
        // Software-pipelined: next chunk's A/B fragments are loaded while the
        // current WMMA runs, so the chain waits on dscnt<=4, not dscnt==0.
        v16bf a0 = *(const v16bf*)(myW);
        v16bf b0 = *(const v16bf*)(hSrc);
        #pragma unroll 8
        for (int ck = 0; ck < KCHUNKS - 1; ++ck) {
            v16bf a1 = *(const v16bf*)(myW + (unsigned)(ck + 1) * 512u);
            v16bf b1 = *(const v16bf*)(hSrc + (ck + 1) * hStride);
            c = __builtin_amdgcn_wmma_f32_16x16x32_bf16(false, a0, false, b0,
                                                        (short)0, c, false, false);
            a0 = a1;
            b0 = b1;
        }
        c = __builtin_amdgcn_wmma_f32_16x16x32_bf16(false, a0, false, b0,
                                                    (short)0, c, false, false);

        // tanh + publish this WG's 64-row h_t chunk (bf16 pairs, ping-pong)
        __bf16* hOut = (t & 1) ? hpong : hping;
        if (inCol0) {
            int r0 = rowBase + halfA * 8;                   // multiple of 8 -> 4B aligned
            unsigned* dst = (unsigned*)(hOut + r0);
            #pragma unroll
            for (int v = 0; v < 4; ++v) {
                union { __bf16 h[2]; unsigned u32; } pk;
                pk.h[0] = (__bf16)tanhf(c[2 * v + 0]);
                pk.h[1] = (__bf16)tanhf(c[2 * v + 1]);
                dst[v] = pk.u32;
            }
        }

        grid_sync(arrive, release, (unsigned)(t + 1));

        // broadcast full h_t back into LDS via the Tensor Data Mover
        if (wave == 0) tdm_load_h(hOut, ldsH);
        __syncthreads();

        // readout y_t = W_out . h_t + b   (WG0 only, deterministic)
        if (wg == 0) {
            float acc = 0.0f;
            for (int j = tid; j < RES; j += TPB)
                acc += ldsWout[j] * (float)ldsH[j];
            #pragma unroll
            for (int o = 16; o > 0; o >>= 1) acc += __shfl_down(acc, o, 32);
            if (lane == 0) ldsRed[wave] = acc;
            __syncthreads();
            if (tid == 0)
                out[t] = ldsRed[0] + ldsRed[1] + ldsRed[2] + ldsRed[3] + bo;
            __syncthreads();
        }
    }
}

// ---------------------------------------------------------------------------
extern "C" void kernel_launch(void* const* d_in, const int* in_sizes, int n_in,
                              void* d_out, int out_size, void* d_ws, size_t ws_size,
                              hipStream_t stream) {
    const float* u     = (const float*)d_in[0];
    const float* W_in  = (const float*)d_in[1];
    const float* W     = (const float*)d_in[2];
    const float* W_out = (const float*)d_in[3];
    const float* b_out = (const float*)d_in[4];
    const float* h0    = (const float*)d_in[5];
    float* out = (float*)d_out;

    char* ws = (char*)d_ws;
    unsigned* counters = (unsigned*)(ws + WS_CNT_OFF);
    __bf16*   hping    = (__bf16*)(ws + WS_HPING_OFF);
    __bf16*   hpong    = (__bf16*)(ws + WS_HPONG_OFF);

    esn_init_kernel<<<(RES + 255) / 256, 256, 0, stream>>>(h0, hping, hpong, counters);

    (void)hipFuncSetAttribute((const void*)esn_persistent_kernel,
                              hipFuncAttributeMaxDynamicSharedMemorySize,
                              (int)LDS_TOTAL_B);

    esn_persistent_kernel<<<NWG, TPB, LDS_TOTAL_B, stream>>>(
        u, W_in, W, W_out, b_out, out, hping, hpong, counters);
}
